// SparseAttentionEdgePredictorLayer_18872086298685
// MI455X (gfx1250) — compile-verified, matrix operand order checked by reading, not run on hardware
//
#include <hip/hip_runtime.h>

#define NODES 8192
#define SIZE  256

typedef unsigned short u16;
typedef unsigned int   u32;
typedef __bf16 bf16_t;
typedef bf16_t v16bf __attribute__((ext_vector_type(16)));
typedef float  v8f   __attribute__((ext_vector_type(8)));
typedef u32    u32x4 __attribute__((ext_vector_type(4)));
typedef float  f32x4 __attribute__((ext_vector_type(4)));

union Frag32B { u32x4 u[2]; v16bf v; };

__device__ __forceinline__ u16 f2bf(float x) {
    u32 u = __float_as_uint(x);
    u32 r = ((u >> 16) & 1u) + 0x7FFFu;   // round-to-nearest-even
    return (u16)((u + r) >> 16);
}

// ---------------- zero the row-sum accumulator ----------------
__global__ void zero_summed(float* __restrict__ s) {
    s[blockIdx.x * 256 + threadIdx.x] = 0.0f;
}

// ---------------- build bf16 A^T / B^T (row-major per node) ----------------
// f is [SIZE][NODES] row-major; we emit At[i][s] = wq[s]*f[s][i], Bt likewise.
__global__ __launch_bounds__(256)
void prep_ab(const float* __restrict__ f, const float* __restrict__ wq,
             const float* __restrict__ wk, u16* __restrict__ At, u16* __restrict__ Bt)
{
    __shared__ float tile[32][33];
    const int tx = threadIdx.x;           // 0..31
    const int ty = threadIdx.y;           // 0..7
    const int i0 = blockIdx.x * 32;
    const int s0 = blockIdx.y * 32;
#pragma unroll
    for (int k = 0; k < 4; ++k) {
        int s = s0 + ty + 8 * k;
        tile[ty + 8 * k][tx] = f[(size_t)s * NODES + i0 + tx];   // coalesced along i
    }
    __syncthreads();
    const float wqv = wq[s0 + tx];
    const float wkv = wk[s0 + tx];
#pragma unroll
    for (int k = 0; k < 4; ++k) {
        int i = i0 + ty + 8 * k;
        float v = tile[tx][ty + 8 * k];                          // transposed read
        At[(size_t)i * SIZE + s0 + tx] = f2bf(wqv * v);          // coalesced along s
        Bt[(size_t)i * SIZE + s0 + tx] = f2bf(wkv * v);
    }
}

// ---------------- WMMA GEMM + exp + masked row-sum ----------------
// One 16x16 output tile per wave; 8 waves/block share the same 16 rows.
__global__ __launch_bounds__(256)
void gemm_exp(const u16* __restrict__ At, const u16* __restrict__ Bt,
              const int* __restrict__ nbr, float* __restrict__ out,
              float* __restrict__ summed)
{
    __shared__ float rowsum[16];
    const int tid  = threadIdx.x;
    const int wave = tid >> 5;
    const int lane = tid & 31;
    const int half = lane >> 4;           // 0: lanes 0-15, 1: lanes 16-31
    const int l    = lane & 15;
    const int m0   = (blockIdx.x >> 6) * 16;
    const int n0   = (blockIdx.x & 63) * 128 + wave * 16;

    if (tid < 16) rowsum[tid] = 0.0f;
    __syncthreads();

    // Per-lane fragment pointers: row-major bf16, 256 elems/row = 32 u32x4/row.
    const u32x4* Ap = (const u32x4*)(At + (size_t)(m0 + l) * SIZE);
    const u32x4* Bp = (const u32x4*)(Bt + (size_t)(n0 + l) * SIZE);

    v8f c = {};
#pragma unroll
    for (int kk = 0; kk < SIZE / 32; ++kk) {
        Frag32B a, b;
        // 16-bit A/B layout: lanes 0-15 K=0-7 & 16-23; lanes 16-31 K=8-15 & 24-31
        a.u[0] = Ap[kk * 4 + half];
        a.u[1] = Ap[kk * 4 + half + 2];
        b.u[0] = Bp[kk * 4 + half];
        b.u[1] = Bp[kk * 4 + half + 2];
        c = __builtin_amdgcn_wmma_f32_16x16x32_bf16(
                false, a.v, false, b.v, (short)0, c, false, false);
    }

#pragma unroll
    for (int r = 0; r < 8; ++r) {
        const int row = m0 + half * 8 + r;    // C VGPR r: M=r (lo half), M=r+8 (hi half)
        const int col = n0 + l;
        const size_t idx = (size_t)row * NODES + col;
        float e = __expf(c[r]);
        out[idx] = e;                          // stash S; divided in a later pass
        float ms = (nbr[idx] != 0) ? e : 0.0f;
#pragma unroll
        for (int off = 1; off < 16; off <<= 1) // reduce over the tile's 16 columns
            ms += __shfl_xor(ms, off, 32);
        if (l == 0) atomicAdd(&rowsum[half * 8 + r], ms);
    }
    __syncthreads();
    if (tid < 16) atomicAdd(&summed[m0 + tid], rowsum[tid]);
}

// ---------------- reciprocal of summed ----------------
__global__ void recip_k(float* __restrict__ s) {
    int i = blockIdx.x * 256 + threadIdx.x;
    s[i] = 1.0f / s[i];
}

// ---------------- out[i][j] *= 1/summed[j] (column-wise) ----------------
__global__ __launch_bounds__(256)
void scale_cols(float* __restrict__ out, const float* __restrict__ rsum)
{
    const int cols4 = NODES / 4;
    const size_t total = (size_t)NODES * cols4;
    f32x4* o = (f32x4*)out;
    const f32x4* rp = (const f32x4*)rsum;
    for (size_t idx = (size_t)blockIdx.x * blockDim.x + threadIdx.x; idx < total;
         idx += (size_t)gridDim.x * blockDim.x) {
        int j4 = (int)(idx % cols4);
        f32x4 v = o[idx];
        f32x4 r = rp[j4];
        o[idx] = v * r;
    }
}

extern "C" void kernel_launch(void* const* d_in, const int* in_sizes, int n_in,
                              void* d_out, int out_size, void* d_ws, size_t ws_size,
                              hipStream_t stream)
{
    const float* f   = (const float*)d_in[0];   // [256][8192]
    const float* wq  = (const float*)d_in[1];   // [256]
    const float* wk  = (const float*)d_in[2];   // [256]
    const int*   nbr = (const int*)d_in[3];     // [8192][8192]
    float* out = (float*)d_out;                 // [8192][8192]

    char* ws = (char*)d_ws;
    float* summed = (float*)ws;                                   // 32 KB
    u16* At = (u16*)(ws + 32768);                                 // 4 MB
    u16* Bt = (u16*)(ws + 32768 + (size_t)NODES * SIZE * sizeof(u16)); // 4 MB

    zero_summed<<<NODES / 256, 256, 0, stream>>>(summed);
    prep_ab<<<dim3(NODES / 32, SIZE / 32), dim3(32, 8), 0, stream>>>(f, wq, wk, At, Bt);
    gemm_exp<<<(NODES / 16) * (NODES / 128), 256, 0, stream>>>(At, Bt, nbr, out, summed);
    recip_k<<<NODES / 256, 256, 0, stream>>>(summed);
    scale_cols<<<4096, 256, 0, stream>>>(out, summed);
}
